// Attention_layer_32762010534061
// MI455X (gfx1250) — matrix operand order repaired and checked by reading, not compiled, not run on hardware
//
#include <hip/hip_runtime.h>

// Problem constants (match reference)
#define Bc 4
#define Cc 64
#define Dc 32
#define Hc 64
#define Wc 64
#define Sc 32
#define PW 16                 // pixels (consecutive w) per workgroup
#define NTHREADS 512          // 16 waves (wave32), one wave per pixel
#define CSTRIDE (Dc*Hc*Wc)    // x stride over c = 131072
#define DSTRIDE (Hc*Wc)       // x stride over d = 4096

// Padded pitches (halves): lane strides in dwords avoid 64-bank conflicts,
// all fragment/store bases stay 16B-aligned (multiples of 8 halves).
#define WPITCH 72             // weight rows of 64 -> 72
#define SPITCH 40             // 32-wide stage-matrix rows -> 40 (also Wo rows)
#define SPP    (32*SPITCH)    // per-pixel stage matrix = 1280 halves
#define XPITCH 2056           // per-pixel X slab: 32*64 + 8

typedef __attribute__((ext_vector_type(16))) _Float16 v16h;
typedef __attribute__((ext_vector_type(8)))  _Float16 v8h;
typedef __attribute__((ext_vector_type(4)))  _Float16 v4h;
typedef __attribute__((ext_vector_type(8)))  float    v8f;

// A-fragment: row-major [m][k] source, k contiguous.
// Per lane: elements 0..7 at (m, kBase+hi*8), elements 8..15 at +16 halves.
__device__ __forceinline__ v16h fragA_ld(const _Float16* base, int idx) {
  const v8h* q = (const v8h*)(base + idx);   // idx is a multiple of 8
  v8h lo = q[0];
  v8h hi = q[2];                             // +16 halves
  return __builtin_shufflevector(lo, hi, 0,1,2,3,4,5,6,7,8,9,10,11,12,13,14,15);
}

// B-fragment: column-major [n][k] source, k contiguous.
// Per lane: elements 0..7 at (n, kBase+hi*8), elements 8..15 at +16 halves.
__device__ __forceinline__ v16h fragB_ld(const _Float16* base, int idx) {
  const v8h* q = (const v8h*)(base + idx);   // idx is a multiple of 8
  v8h lo = q[0];
  v8h hi = q[2];                             // +16 halves
  return __builtin_shufflevector(lo, hi, 0,1,2,3,4,5,6,7,8,9,10,11,12,13,14,15);
}

#define WMMA(acc, a, b) \
  __builtin_amdgcn_wmma_f32_16x16x32_f16(false, (a), false, (b), (short)0, (acc), false, false)

__global__ __launch_bounds__(NTHREADS, 1)
void attn3d_fused_kernel(const float* __restrict__ x,
                         const float* __restrict__ Wk, const float* __restrict__ bk,
                         const float* __restrict__ Wq, const float* __restrict__ bq,
                         const float* __restrict__ Wv, const float* __restrict__ bv,
                         const float* __restrict__ Wo, const float* __restrict__ bo,
                         float* __restrict__ out) {
  __shared__ __align__(16) _Float16 sWk[Sc * WPITCH];   // [s][c] row-major (A)
  __shared__ __align__(16) _Float16 sWq[Sc * WPITCH];   // pre-scaled by 1/sqrt(S)
  __shared__ __align__(16) _Float16 sWv[Sc * WPITCH];   // [s][c]; read as col-major B for V^T
  __shared__ __align__(16) _Float16 sWo[Cc * SPITCH];   // [c][s] row-major (A)
  __shared__ float sbk[Sc], sbq[Sc], sbv[Sc], sbo[Cc];
  __shared__ __align__(16) _Float16 sXb[PW * XPITCH];   // [p][i][c]; reused as out tile [c][d][p]
  __shared__ __align__(16) _Float16 sStage[3 * PW * SPP];

  _Float16* sKt = sStage;                // [p](Kt)[i][s] row-major  (A for scores)
  _Float16* sQc = sStage + PW * SPP;     // [p](Q) [j][s] col-major  (B for scores)
  _Float16* sV  = sStage + 2 * PW * SPP; // [p](V) [s][i] row-major  (A for O)
  _Float16* sAc = sKt;                   // [p](A) [j][i] col-major  (B for O)      (alias, per-pixel private)
  _Float16* sOc = sQc;                   // [p](O) [j][s] col-major  (B for out)    (alias, per-pixel private)

  const int t    = threadIdx.x;
  const int lane = t & 31;
  const int p    = t >> 5;               // wave id == pixel within w-tile
  const int nn   = lane & 15;
  const int hi   = lane >> 4;
  const int hi8  = hi * 8;
  const int hi16 = hi * 16;
  const int pS   = p * SPP;              // per-pixel stage offset
  const int pX   = p * XPITCH;           // per-pixel X offset

  const int wblk = blockIdx.x & 3;       // W / PW = 4
  const int h    = (blockIdx.x >> 2) & 63;
  const int b    = blockIdx.x >> 8;
  const int w0   = wblk * PW;
  const long gbase = (long)b * Cc * CSTRIDE + (long)h * Wc + w0;

  const float invSqrtS = 0.1767766952966369f;  // 1/sqrt(S), folded into staged Wq/bq

  // ---------------- Phase 0: stage weights + X tile ----------------
  for (int i = t; i < Sc * Cc; i += NTHREADS) {
    const int r = i >> 6, cI = i & 63;         // weight rows of 64
    sWk[r * WPITCH + cI] = (_Float16)Wk[i];
    sWq[r * WPITCH + cI] = (_Float16)(Wq[i] * invSqrtS);
    sWv[r * WPITCH + cI] = (_Float16)Wv[i];
    const int r2 = i >> 5, c2 = i & 31;        // Wo rows of 32
    sWo[r2 * SPITCH + c2] = (_Float16)Wo[i];
  }
  if (t < Sc) { sbk[t] = bk[t]; sbq[t] = bq[t] * invSqrtS; sbv[t] = bv[t]; }
  if (t < Cc) { sbo[t] = bo[t]; }
  // X: float4 coalesced global loads, transposing scatter into [p][i][c]
  for (int q = t; q < Cc * Dc * (PW / 4); q += NTHREADS) {
    const int wq = q & 3;
    const int d  = (q >> 2) & (Dc - 1);
    const int c  = q >> 7;
    const long g = gbase + (long)c * CSTRIDE + d * DSTRIDE + wq * 4;
    const float4 xr = *(const float4*)(x + g);
    const int p0 = wq * 4;
    sXb[(p0 + 0) * XPITCH + d * Cc + c] = (_Float16)xr.x;
    sXb[(p0 + 1) * XPITCH + d * Cc + c] = (_Float16)xr.y;
    sXb[(p0 + 2) * XPITCH + d * Cc + c] = (_Float16)xr.z;
    sXb[(p0 + 3) * XPITCH + d * Cc + c] = (_Float16)xr.w;
  }
  __syncthreads();

  // ---------------- Phase 1a: K and Q projections (m=s, n=i/j) ----------------
#pragma unroll
  for (int pr = 0; pr < 2; ++pr) {
    const _Float16* w  = (pr == 0) ? sWk : sWq;
    const float* bias  = (pr == 0) ? sbk : sbq;
    float bReg[2][8];
#pragma unroll
    for (int mt = 0; mt < 2; ++mt)
#pragma unroll
      for (int r = 0; r < 8; ++r) bReg[mt][r] = bias[mt * 16 + r + hi8];
#pragma unroll
    for (int mt = 0; mt < 2; ++mt)
#pragma unroll
      for (int nt = 0; nt < 2; ++nt) {
        v8f acc = {};
#pragma unroll
        for (int kc = 0; kc < 2; ++kc) {
          v16h a  = fragA_ld(w,   (mt * 16 + nn) * WPITCH + kc * 32 + hi8);
          v16h bb = fragB_ld(sXb, pX + (nt * 16 + nn) * Cc + kc * 32 + hi8);
          acc = WMMA(acc, a, bb);
        }
        const int ii = nt * 16 + nn;          // depth position (i or j)
        v8h pk;
#pragma unroll
        for (int r = 0; r < 8; ++r) pk[r] = (_Float16)(acc[r] + bReg[mt][r]);
        _Float16* dst = ((pr == 0) ? sKt : sQc) + pS + ii * SPITCH + mt * 16 + hi8;
        *(v8h*)dst = pk;                      // one ds_store_b128: [n][m] contiguous
      }
  }

  // ---------------- Phase 1b: V^T projection (m=i, n=s): Vt = X^T Wv^T ----------------
  {
    const float bv0 = sbv[nn];
    const float bv1 = sbv[16 + nn];
#pragma unroll
    for (int mt = 0; mt < 2; ++mt)          // i tiles
#pragma unroll
      for (int ns = 0; ns < 2; ++ns) {      // s tiles
        v8f acc = {};
#pragma unroll
        for (int kc = 0; kc < 2; ++kc) {
          v16h a  = fragA_ld(sXb, pX + (mt * 16 + nn) * Cc + kc * 32 + hi8);
          v16h bb = fragB_ld(sWv, (ns * 16 + nn) * WPITCH + kc * 32 + hi8);
          acc = WMMA(acc, a, bb);
        }
        const float bvv = ns ? bv1 : bv0;   // bias depends on n=s only
        const int s = ns * 16 + nn;
        v8h pk;
#pragma unroll
        for (int r = 0; r < 8; ++r) pk[r] = (_Float16)(acc[r] + bvv);
        *(v8h*)(&sV[pS + s * SPITCH + mt * 16 + hi8]) = pk;  // V[s][i], i contiguous
      }
  }
  __syncthreads();  // all waves done reading sXb (it becomes the out tile)

  // ---------------- Phase 2: scores = K^T Q ----------------
  v8f st[2][2];
#pragma unroll
  for (int mt = 0; mt < 2; ++mt)
#pragma unroll
    for (int nt = 0; nt < 2; ++nt) {
      v8f acc = {};
      v16h a  = fragA_ld(sKt, pS + (mt * 16 + nn) * SPITCH + hi8);
      v16h bb = fragB_ld(sQc, pS + (nt * 16 + nn) * SPITCH + hi8);
      st[mt][nt] = WMMA(acc, a, bb);
    }

  // ---------------- Phase 3: softmax over i (rows) per column j ----------------
#pragma unroll
  for (int nt = 0; nt < 2; ++nt) {
    float mx = -3.0e38f;
#pragma unroll
    for (int mt = 0; mt < 2; ++mt)
#pragma unroll
      for (int r = 0; r < 8; ++r) mx = fmaxf(mx, st[mt][nt][r]);
    mx = fmaxf(mx, __shfl_xor(mx, 16, 32));   // partner lane holds other 16 rows
    float sum = 0.f;
#pragma unroll
    for (int mt = 0; mt < 2; ++mt)
#pragma unroll
      for (int r = 0; r < 8; ++r) {
        const float e = __expf(st[mt][nt][r] - mx);
        st[mt][nt][r] = e;
        sum += e;
      }
    sum += __shfl_xor(sum, 16, 32);
    const float inv = 1.0f / sum;
    const int j = nt * 16 + nn;
#pragma unroll
    for (int mt = 0; mt < 2; ++mt) {
      v8h pk;
#pragma unroll
      for (int r = 0; r < 8; ++r) pk[r] = (_Float16)(st[mt][nt][r] * inv);
      *(v8h*)(&sAc[pS + j * SPITCH + mt * 16 + hi8]) = pk;   // A[j][i], i contiguous
    }
  }

  // ---------------- Phase 4: O = V @ A (m=s, k=i, n=j) ----------------
#pragma unroll
  for (int mt = 0; mt < 2; ++mt)
#pragma unroll
    for (int nt = 0; nt < 2; ++nt) {
      v8f acc = {};
      v16h a  = fragA_ld(sV,  pS + (mt * 16 + nn) * SPITCH + hi8);
      v16h bb = fragB_ld(sAc, pS + (nt * 16 + nn) * SPITCH + hi8);
      acc = WMMA(acc, a, bb);
      const int j = nt * 16 + nn;
      v8h pk;
#pragma unroll
      for (int r = 0; r < 8; ++r) pk[r] = (_Float16)acc[r];
      *(v8h*)(&sOc[pS + j * SPITCH + mt * 16 + hi8]) = pk;   // O[j][s], s contiguous
    }

  // ---------------- Phase 5: Out = Wo @ O + bo -> staged into sXb as [c][d][p] ----------------
#pragma unroll
  for (int mt = 0; mt < 4; ++mt) {
    float bReg[8];
#pragma unroll
    for (int r = 0; r < 8; ++r) bReg[r] = sbo[mt * 16 + r + hi8];
#pragma unroll
    for (int nt = 0; nt < 2; ++nt) {
      v8f acc = {};
      v16h a  = fragA_ld(sWo, (mt * 16 + nn) * SPITCH + hi8);
      v16h bb = fragB_ld(sOc, pS + (nt * 16 + nn) * SPITCH + hi8);
      acc = WMMA(acc, a, bb);
#pragma unroll
      for (int r = 0; r < 8; ++r) {
        const int c = mt * 16 + r + hi8;
        const int d = nt * 16 + nn;
        sXb[(c * Dc + d) * PW + p] = (_Float16)(acc[r] + bReg[r]);  // scattered by design (p-major out tile)
      }
    }
  }
  __syncthreads();

  // ---------------- Phase 6: residual add + float4 coalesced write ----------------
  for (int q = t; q < Cc * Dc * (PW / 4); q += NTHREADS) {
    const int wq = q & 3;
    const int d  = (q >> 2) & (Dc - 1);
    const int c  = q >> 7;
    const long g = gbase + (long)c * CSTRIDE + d * DSTRIDE + wq * 4;
    const float4 xr = *(const float4*)(x + g);   // L2-resident re-read (128MB x < 192MB L2)
    const v4h  o  = *(const v4h*)(&sXb[(c * Dc + d) * PW + wq * 4]);
    float4 r;
    r.x = xr.x + (float)o[0];
    r.y = xr.y + (float)o[1];
    r.z = xr.z + (float)o[2];
    r.w = xr.w + (float)o[3];
    *(float4*)(out + g) = r;
  }
}

extern "C" void kernel_launch(void* const* d_in, const int* in_sizes, int n_in,
                              void* d_out, int out_size, void* d_ws, size_t ws_size,
                              hipStream_t stream) {
  (void)in_sizes; (void)n_in; (void)d_ws; (void)ws_size; (void)out_size;
  const float* x  = (const float*)d_in[0];
  const float* Wk = (const float*)d_in[1];
  const float* bk = (const float*)d_in[2];
  const float* Wq = (const float*)d_in[3];
  const float* bq = (const float*)d_in[4];
  const float* Wv = (const float*)d_in[5];
  const float* bv = (const float*)d_in[6];
  const float* Wo = (const float*)d_in[7];
  const float* bo = (const float*)d_in[8];
  float* out = (float*)d_out;

  dim3 grid(Bc * Hc * (Wc / PW));   // 4 * 64 * 4 = 1024 workgroups
  attn3d_fused_kernel<<<grid, NTHREADS, 0, stream>>>(x, Wk, bk, Wq, bq, Wv, bv, Wo, bo, out);
}
